// SparseResBlock_25598005084683
// MI455X (gfx1250) — matrix operand order repaired and checked by reading, not compiled, not run on hardware
//
#include <hip/hip_runtime.h>

// ---------------------------------------------------------------------------
// Types
// ---------------------------------------------------------------------------
typedef __bf16 bf16;
typedef __attribute__((ext_vector_type(16))) __bf16       v16bf;
typedef __attribute__((ext_vector_type(8)))  float        v8f;
typedef __attribute__((ext_vector_type(4)))  unsigned int u32x4;
typedef __attribute__((ext_vector_type(8)))  __bf16       bf16x8;
typedef __attribute__((ext_vector_type(4)))  __bf16       bf16x4;

union FragBF { v16bf v; u32x4 q[2]; };

constexpr int N_PTS       = 1000000;
constexpr int C           = 64;
constexpr int STAT_BLOCKS = 1024;
constexpr int WSZ         = 9 * 64 * 64;   // weights per layer (elements)

// ---------------------------------------------------------------------------
// Init: zero stats accumulators and the zero-pad row (index N) of each buffer
// ---------------------------------------------------------------------------
__global__ void k_init(float* __restrict__ stats,
                       bf16* __restrict__ b0, bf16* __restrict__ b1,
                       bf16* __restrict__ b2) {
  int t = threadIdx.x;
  if (t < 3 * 128) stats[t] = 0.0f;
  if (t < 64) {
    size_t p = (size_t)N_PTS * C + t;
    b0[p] = (bf16)0.0f;
    b1[p] = (bf16)0.0f;
    b2[p] = (bf16)0.0f;
  }
}

// ---------------------------------------------------------------------------
// Cast fp32 features -> bf16 (vectorized 4-wide)
// ---------------------------------------------------------------------------
__global__ __launch_bounds__(256)
void k_cast(const float* __restrict__ f, bf16* __restrict__ o) {
  size_t stride = (size_t)gridDim.x * blockDim.x;
  size_t n4 = (size_t)N_PTS * C / 4;
  for (size_t i = (size_t)blockIdx.x * blockDim.x + threadIdx.x; i < n4; i += stride) {
    float4 v = ((const float4*)f)[i];
    bf16x4 r;
    r[0] = (bf16)v.x; r[1] = (bf16)v.y; r[2] = (bf16)v.z; r[3] = (bf16)v.w;
    ((bf16x4*)o)[i] = r;
  }
}

// ---------------------------------------------------------------------------
// Weight prep: W[t][cin][cout] fp32 -> wt[t][cout][cin] bf16  (B-matrix layout:
// one column (cout) per lane, contiguous cin = K)
// ---------------------------------------------------------------------------
__global__ __launch_bounds__(256)
void k_prepw(const float* __restrict__ w, bf16* __restrict__ wt) {
  int id = blockIdx.x * blockDim.x + threadIdx.x;
  if (id >= WSZ) return;
  int t = id >> 12, rem = id & 4095, o = rem >> 6, i = rem & 63;
  wt[id] = (bf16)w[(t << 12) + (i << 6) + o];
}

// ---------------------------------------------------------------------------
// Conv: out[i] = sum_k feat[nbr[i,k]] @ W[k], bf16 in / bf16 out (pre-norm),
// fp32 WMMA accumulation. One wave = 32 rows x 64 channels (two 16-row
// M-tiles sharing every B-fragment). All gather/store addresses are
// 32-bit voffsets from a uniform SGPR base (GVS mode) to kill the 64-bit
// per-row address chains.
// ---------------------------------------------------------------------------
__global__ __launch_bounds__(256)
void k_conv(const bf16* __restrict__ src,   // (N+1) x 64, row N = zeros
            const bf16* __restrict__ wt,    // [9][64 cout][64 cin]
            const int*  __restrict__ nbr,   // [N][9], values in [0, N]
            bf16* __restrict__ dst) {       // N x 64
  extern __shared__ bf16 lw[];              // 9*64*64 bf16 = 73728 B
  {
    const u32x4* g = (const u32x4*)wt;
    u32x4* s = (u32x4*)lw;
    for (int i = threadIdx.x; i < WSZ / 8; i += 256) s[i] = g[i];
  }
  __syncthreads();

  const int wave = threadIdx.x >> 5;
  const int lane = threadIdx.x & 31;
  const int half = lane >> 4;     // 0: lanes 0-15, 1: lanes 16-31
  const int l15  = lane & 15;
  const long m0  = ((long)blockIdx.x * 8 + wave) * 32;
  if (m0 >= N_PTS) return;        // uniform per wave; no barriers after this

  const int rowA  = (int)m0 + l15;            // M-tile A rows [m0, m0+16)
  const int rowB  = (int)m0 + 16 + l15;       // M-tile B rows [m0+16, m0+32)
  const int rowcA = rowA < N_PTS ? rowA : N_PTS - 1;
  const int rowcB = rowB < N_PTS ? rowB : N_PTS - 1;

  const char* sp = (const char*)src;          // uniform base, 32-bit voffsets
  char*       dp = (char*)dst;
  const unsigned cbB = (unsigned)half * 16u;  // byte offset of lane's K-chunk

  v8f aA0 = {}, aA1 = {}, aA2 = {}, aA3 = {};   // acc for M-tile A, nt 0..3
  v8f aB0 = {}, aB1 = {}, aB2 = {}, aB3 = {};   // acc for M-tile B, nt 0..3

#pragma unroll
  for (int k = 0; k < 9; ++k) {
    const int idx0 = nbr[rowcA * 9 + k];        // N = zero-pad row
    const int idx1 = nbr[rowcB * 9 + k];
    const unsigned o0 = ((unsigned)idx0 << 7) + cbB;   // row byte offset
    const unsigned o1 = ((unsigned)idx1 << 7) + cbB;

    // A fragments (16x32 bf16 each): K chunks of 8 split across lane halves
    FragBF fA0, fA1, fB0, fB1;                  // {tile}{K-slice}
    fA0.q[0] = *(const u32x4*)(sp + o0);
    fA0.q[1] = *(const u32x4*)(sp + o0 + 32);
    fA1.q[0] = *(const u32x4*)(sp + o0 + 64);
    fA1.q[1] = *(const u32x4*)(sp + o0 + 96);
    fB0.q[0] = *(const u32x4*)(sp + o1);
    fB0.q[1] = *(const u32x4*)(sp + o1 + 32);
    fB1.q[0] = *(const u32x4*)(sp + o1 + 64);
    fB1.q[1] = *(const u32x4*)(sp + o1 + 96);

    const bf16* wk = lw + k * 64 * 64;
    const int kb = half * 8;                    // B: K chunks of 8 split across lane halves
#pragma unroll
    for (int nt = 0; nt < 4; ++nt) {
      const bf16* wrow = wk + (nt * 16 + l15) * 64;
      v8f& accA = nt == 0 ? aA0 : nt == 1 ? aA1 : nt == 2 ? aA2 : aA3;
      v8f& accB = nt == 0 ? aB0 : nt == 1 ? aB1 : nt == 2 ? aB2 : aB3;

      FragBF b0;                                // K-slice 0 weights
      b0.q[0] = *(const u32x4*)(wrow + kb);
      b0.q[1] = *(const u32x4*)(wrow + kb + 16);
      accA = __builtin_amdgcn_wmma_f32_16x16x32_bf16(false, fA0.v, false, b0.v,
                                                     (short)0, accA, false, false);
      accB = __builtin_amdgcn_wmma_f32_16x16x32_bf16(false, fB0.v, false, b0.v,
                                                     (short)0, accB, false, false);

      FragBF b1;                                // K-slice 1 weights
      b1.q[0] = *(const u32x4*)(wrow + 32 + kb);
      b1.q[1] = *(const u32x4*)(wrow + 32 + kb + 16);
      accA = __builtin_amdgcn_wmma_f32_16x16x32_bf16(false, fA1.v, false, b1.v,
                                                     (short)0, accA, false, false);
      accB = __builtin_amdgcn_wmma_f32_16x16x32_bf16(false, fB1.v, false, b1.v,
                                                     (short)0, accB, false, false);
    }
  }

  // C/D layout: VGPR j holds M=j (lanes 0-15) / M=j+8 (lanes 16-31), N=lane&15
#pragma unroll
  for (int j = 0; j < 8; ++j) {
    int r0 = (int)m0 + half * 8 + j;
    if (r0 < N_PTS) {
      unsigned doff = ((unsigned)r0 << 7) + (unsigned)l15 * 2u;
      *(bf16*)(dp + doff)      = (bf16)aA0[j];
      *(bf16*)(dp + doff + 32) = (bf16)aA1[j];
      *(bf16*)(dp + doff + 64) = (bf16)aA2[j];
      *(bf16*)(dp + doff + 96) = (bf16)aA3[j];
    }
    int r1 = (int)m0 + 16 + half * 8 + j;
    if (r1 < N_PTS) {
      unsigned doff = ((unsigned)r1 << 7) + (unsigned)l15 * 2u;
      *(bf16*)(dp + doff)      = (bf16)aB0[j];
      *(bf16*)(dp + doff + 32) = (bf16)aB1[j];
      *(bf16*)(dp + doff + 64) = (bf16)aB2[j];
      *(bf16*)(dp + doff + 96) = (bf16)aB3[j];
    }
  }
}

// ---------------------------------------------------------------------------
// Per-block partial sums / sums-of-squares per channel (deterministic order)
// ---------------------------------------------------------------------------
__global__ __launch_bounds__(256)
void k_stats(const bf16* __restrict__ y, float* __restrict__ part) {
  __shared__ float ls[2][64];
  const int c = threadIdx.x & 63;
  const int g = threadIdx.x >> 6;   // 0..3
  float s = 0.0f, s2 = 0.0f;
  for (long r = (long)blockIdx.x * 4 + g; r < N_PTS; r += (long)gridDim.x * 4) {
    float v = (float)y[(size_t)r * C + c];
    s += v; s2 += v * v;
  }
  if (g == 0) { ls[0][c] = s;  ls[1][c] = s2; } __syncthreads();
  if (g == 1) { ls[0][c] += s; ls[1][c] += s2; } __syncthreads();
  if (g == 2) { ls[0][c] += s; ls[1][c] += s2; } __syncthreads();
  if (g == 3) { ls[0][c] += s; ls[1][c] += s2; } __syncthreads();
  if (threadIdx.x < 128)
    part[(size_t)blockIdx.x * 128 + threadIdx.x] = ls[threadIdx.x >> 6][threadIdx.x & 63];
}

__global__ void k_reduce(const float* __restrict__ part, float* __restrict__ stats) {
  int t = threadIdx.x;  // 128
  float a = 0.0f;
  for (int b = 0; b < STAT_BLOCKS; ++b) a += part[b * 128 + t];
  stats[t] = a;         // [0..63]=sum, [64..127]=sumsq
}

// ---------------------------------------------------------------------------
// BN (eps 1e-4) + LeakyReLU(0.333), bf16 -> bf16
// ---------------------------------------------------------------------------
__global__ __launch_bounds__(256)
void k_bn(const bf16* __restrict__ y, const float* __restrict__ stats,
          const float* __restrict__ gamma, const float* __restrict__ beta,
          bf16* __restrict__ x) {
  __shared__ float sa[64], sb[64];
  if (threadIdx.x < 64) {
    int c = threadIdx.x;
    float mu  = stats[c] * (1.0f / N_PTS);
    float var = stats[64 + c] * (1.0f / N_PTS) - mu * mu;
    float a   = rsqrtf(var + 1e-4f) * gamma[c];
    sa[c] = a; sb[c] = beta[c] - mu * a;
  }
  __syncthreads();
  size_t stride = (size_t)gridDim.x * blockDim.x;
  size_t n8 = (size_t)N_PTS * C / 8;
  for (size_t i = (size_t)blockIdx.x * blockDim.x + threadIdx.x; i < n8; i += stride) {
    bf16x8 v = ((const bf16x8*)y)[i];
    int c0 = (int)(i & 7) * 8;
    bf16x8 r;
#pragma unroll
    for (int j = 0; j < 8; ++j) {
      float f = (float)v[j] * sa[c0 + j] + sb[c0 + j];
      r[j] = (bf16)(f >= 0.0f ? f : 0.333f * f);
    }
    ((bf16x8*)x)[i] = r;
  }
}

// ---------------------------------------------------------------------------
// InstanceNorm (eps 1e-5) + residual + LeakyReLU(0.01) -> fp32 output
// ---------------------------------------------------------------------------
__global__ __launch_bounds__(256)
void k_final(const bf16* __restrict__ y, const float* __restrict__ stats,
             const bf16* __restrict__ res, float* __restrict__ out) {
  __shared__ float sa[64], sb[64];
  if (threadIdx.x < 64) {
    int c = threadIdx.x;
    float mu  = stats[c] * (1.0f / N_PTS);
    float var = stats[64 + c] * (1.0f / N_PTS) - mu * mu;
    float rs  = rsqrtf(var + 1e-5f);
    sa[c] = rs; sb[c] = -mu * rs;
  }
  __syncthreads();
  size_t stride = (size_t)gridDim.x * blockDim.x;
  size_t n4 = (size_t)N_PTS * C / 4;
  for (size_t i = (size_t)blockIdx.x * blockDim.x + threadIdx.x; i < n4; i += stride) {
    bf16x4 v = ((const bf16x4*)y)[i];
    bf16x4 rr = ((const bf16x4*)res)[i];
    int c0 = (int)(i & 15) * 4;
    float4 o;
    float z;
    z = (float)v[0] * sa[c0 + 0] + sb[c0 + 0] + (float)rr[0]; o.x = z >= 0.0f ? z : 0.01f * z;
    z = (float)v[1] * sa[c0 + 1] + sb[c0 + 1] + (float)rr[1]; o.y = z >= 0.0f ? z : 0.01f * z;
    z = (float)v[2] * sa[c0 + 2] + sb[c0 + 2] + (float)rr[2]; o.z = z >= 0.0f ? z : 0.01f * z;
    z = (float)v[3] * sa[c0 + 3] + sb[c0 + 3] + (float)rr[3]; o.w = z >= 0.0f ? z : 0.01f * z;
    ((float4*)out)[i] = o;
  }
}

// ---------------------------------------------------------------------------
// Host-side orchestration
// ---------------------------------------------------------------------------
extern "C" void kernel_launch(void* const* d_in, const int* in_sizes, int n_in,
                              void* d_out, int out_size, void* d_ws, size_t ws_size,
                              hipStream_t stream) {
  (void)in_sizes; (void)n_in; (void)out_size; (void)ws_size;
  const float* feat = (const float*)d_in[0];
  const int*   nbr  = (const int*)d_in[1];
  const float* W1   = (const float*)d_in[2];
  const float* g1   = (const float*)d_in[3];
  const float* b1   = (const float*)d_in[4];
  const float* W2   = (const float*)d_in[5];
  const float* g2   = (const float*)d_in[6];
  const float* b2   = (const float*)d_in[7];
  const float* W3   = (const float*)d_in[8];
  float* out = (float*)d_out;

  char* ws = (char*)d_ws;
  float* stats = (float*)ws;                                   // 3 x 128 f32
  float* part  = (float*)(ws + 2048);                          // 1024 x 128 f32
  bf16*  wt    = (bf16*)(ws + 2048 + (size_t)STAT_BLOCKS * 128 * 4);
  size_t off   = 2048 + (size_t)STAT_BLOCKS * 128 * 4 + (size_t)3 * WSZ * 2;
  off = (off + 255) & ~(size_t)255;
  size_t szBuf = (((size_t)(N_PTS + 1) * C * 2) + 255) & ~(size_t)255;
  bf16* B0 = (bf16*)(ws + off);
  bf16* B1 = (bf16*)(ws + off + szBuf);
  bf16* B2 = (bf16*)(ws + off + 2 * szBuf);
  bf16 *wt0 = wt, *wt1 = wt + WSZ, *wt2 = wt + 2 * WSZ;

  k_init<<<1, 512, 0, stream>>>(stats, B0, B1, B2);
  k_cast<<<4096, 256, 0, stream>>>(feat, B0);
  int wb = (WSZ + 255) / 256;
  k_prepw<<<wb, 256, 0, stream>>>(W1, wt0);
  k_prepw<<<wb, 256, 0, stream>>>(W2, wt1);
  k_prepw<<<wb, 256, 0, stream>>>(W3, wt2);

  const int convBlocks = (N_PTS + 255) / 256;   // 8 waves x 32 rows per block
  const size_t lds = (size_t)WSZ * 2;           // 73728 B dynamic LDS (WGP: 320 KB)

  // Layer 1: y1 = conv(feat); x1 = bn_lrelu(y1)  (x1 kept in B2 as residual)
  k_conv<<<convBlocks, 256, lds, stream>>>(B0, wt0, nbr, B1);
  k_stats<<<STAT_BLOCKS, 256, 0, stream>>>(B1, part);
  k_reduce<<<1, 128, 0, stream>>>(part, stats);
  k_bn<<<4096, 256, 0, stream>>>(B1, stats, g1, b1, B2);

  // Layer 2: y2 = conv(x1); x2 = bn_lrelu(y2)  (B0 reused for x2)
  k_conv<<<convBlocks, 256, lds, stream>>>(B2, wt1, nbr, B1);
  k_stats<<<STAT_BLOCKS, 256, 0, stream>>>(B1, part);
  k_reduce<<<1, 128, 0, stream>>>(part, stats + 128);
  k_bn<<<4096, 256, 0, stream>>>(B1, stats + 128, g2, b2, B0);

  // Layer 3: y3 = conv(x2); out = lrelu(inorm(y3) + x1)
  k_conv<<<convBlocks, 256, lds, stream>>>(B0, wt2, nbr, B1);
  k_stats<<<STAT_BLOCKS, 256, 0, stream>>>(B1, part);
  k_reduce<<<1, 128, 0, stream>>>(part, stats + 256);
  k_final<<<8192, 256, 0, stream>>>(B1, stats + 256, B2, out);
}
